// NATLayer_43954695307302
// MI455X (gfx1250) — matrix operand order repaired and checked by reading, not compiled, not run on hardware
//
#include <hip/hip_runtime.h>

// ---------------------------------------------------------------------------
// Types
// ---------------------------------------------------------------------------
typedef __attribute__((ext_vector_type(16))) __bf16 v16bf;
typedef __attribute__((ext_vector_type(8)))  float  v8f;
typedef unsigned short ushort_t;

struct alignas(16) U4 { unsigned x, y, z, w; };
struct alignas(16) Pair { U4 lo, hi; };   // 32 bytes == sizeof(v16bf)

__device__ inline ushort_t f2bf(float f) {
    unsigned u = __builtin_bit_cast(unsigned, f);
    unsigned r = u + 0x7FFFu + ((u >> 16) & 1u);   // round-to-nearest-even
    return (ushort_t)(r >> 16);
}

// branch-free tanh via fast exp (v_exp_f32)
__device__ inline float fast_tanh(float x) {
    float ax = fabsf(x);
    float e  = __expf(-2.0f * ax);
    float t  = (1.0f - e) / (1.0f + e);
    return copysignf(t, x);
}

__device__ inline float gelu_f(float x) {
    const float c = 0.7978845608028654f;           // sqrt(2/pi)
    float t = fast_tanh(c * (x + 0.044715f * x * x * x));
    return 0.5f * x * (1.0f + t);
}

// wave32 all-lanes sum via ds_swizzle XOR butterflies (group-of-32 mode)
__device__ inline float wave_sum32(float x) {
    x += __builtin_bit_cast(float, __builtin_amdgcn_ds_swizzle(__builtin_bit_cast(int, x), 0x041F)); // xor 1
    x += __builtin_bit_cast(float, __builtin_amdgcn_ds_swizzle(__builtin_bit_cast(int, x), 0x081F)); // xor 2
    x += __builtin_bit_cast(float, __builtin_amdgcn_ds_swizzle(__builtin_bit_cast(int, x), 0x101F)); // xor 4
    x += __builtin_bit_cast(float, __builtin_amdgcn_ds_swizzle(__builtin_bit_cast(int, x), 0x201F)); // xor 8
    x += __builtin_bit_cast(float, __builtin_amdgcn_ds_swizzle(__builtin_bit_cast(int, x), 0x401F)); // xor 16
    return x;
}

// ---------------------------------------------------------------------------
// CDNA5 async global -> LDS copy (ASYNCcnt-tracked, 16 bytes per lane)
// ---------------------------------------------------------------------------
__device__ inline void async_copy_b128(const void* gsrc, void* lds_dst) {
    unsigned           loff = (unsigned)(size_t)lds_dst;        // low 32 bits = LDS offset
    unsigned long long gadr = (unsigned long long)(size_t)gsrc;
    asm volatile("global_load_async_to_lds_b128 %0, %1, off"
                 :: "v"(loff), "v"(gadr) : "memory");
}
__device__ inline void wait_async_all() {
    asm volatile("s_wait_asynccnt 0x0" ::: "memory");
}

// ---------------------------------------------------------------------------
// fp32 -> bf16 convert (weights)
// ---------------------------------------------------------------------------
__global__ __launch_bounds__(256) void cvt_bf16_kernel(const float* __restrict__ src,
                                                       ushort_t* __restrict__ dst, int n) {
    int i = blockIdx.x * 256 + threadIdx.x;
    if (i < n) dst[i] = f2bf(src[i]);
}

// ---------------------------------------------------------------------------
// LayerNorm (row length fixed = 512) -> bf16
// ---------------------------------------------------------------------------
__global__ __launch_bounds__(256) void ln_bf16_kernel(const float* __restrict__ in,
                                                      const float* __restrict__ g,
                                                      const float* __restrict__ be,
                                                      ushort_t* __restrict__ out) {
    int row = blockIdx.x;
    int t = threadIdx.x;
    const float* r = in + (size_t)row * 512;
    float x0 = r[t], x1 = r[t + 256];

    __shared__ float ssum[256];
    __shared__ float ssq[256];
    ssum[t] = x0 + x1;
    ssq[t]  = x0 * x0 + x1 * x1;
    __syncthreads();
    for (int s = 128; s > 0; s >>= 1) {
        if (t < s) { ssum[t] += ssum[t + s]; ssq[t] += ssq[t + s]; }
        __syncthreads();
    }
    float mean = ssum[0] * (1.0f / 512.0f);
    float var  = ssq[0] * (1.0f / 512.0f) - mean * mean;
    float rstd = rsqrtf(var + 1e-5f);

    ushort_t* o = out + (size_t)row * 512;
    o[t]       = f2bf((x0 - mean) * rstd * g[t]       + be[t]);
    o[t + 256] = f2bf((x1 - mean) * rstd * g[t + 256] + be[t + 256]);
}

// ---------------------------------------------------------------------------
// bf16 WMMA GEMM: C[M,N] = A[M,K] * B[K,N] + bias, optional residual / GELU
// block tile 128x128, BK=32 (one WMMA K-step), 256 threads = 8 waves (32x64 each)
// Double-buffered LDS filled with GLOBAL_LOAD_ASYNC_TO_LDS_B128.
// mode 0: outF = val ; mode 1: outF = val + resid ; mode 2: outB = bf16(gelu(val))
// ---------------------------------------------------------------------------
#define BM 128
#define BN 128
#define BK 32

__global__ __launch_bounds__(256) void gemm_wmma_kernel(
    const ushort_t* __restrict__ A, const ushort_t* __restrict__ Bw,
    const float* __restrict__ bias, const float* __restrict__ resid,
    float* __restrict__ outF, ushort_t* __restrict__ outB,
    int M, int N, int Kd, int mode)
{
    __shared__ ushort_t As[2][BM][40];    // +8 pad, row = 80B (16B-aligned)
    __shared__ ushort_t Bs[2][BK][136];   // +8 pad, row = 272B (16B-aligned)

    const int t = threadIdx.x;
    const int wave = t >> 5, lane = t & 31;
    const int laneLo = lane & 15, laneHi = lane >> 4;
    const int waveM = wave & 3, waveN = wave >> 2;   // 4 x 2 wave grid
    const int mBase = blockIdx.x * BM;
    const int nBase = blockIdx.y * BN;

    v8f acc[8] = {};   // [m(2)][n(4)]

    // per-thread copy slots: 2 x 16B of A, 2 x 16B of B per tile
    const int aRow0 = t >> 2,          aSeg0 = t & 3;
    const int aRow1 = (t + 256) >> 2,  aSeg1 = (t + 256) & 3;
    const int bRow0 = t >> 4,          bSeg0 = t & 15;
    const int bRow1 = (t + 256) >> 4,  bSeg1 = (t + 256) & 15;

    const int nK = Kd / BK;
    int buf = 0;

    // preload tile 0 into buffer 0
    {
        const int k0 = 0;
        async_copy_b128(A + (size_t)(mBase + aRow0) * Kd + k0 + aSeg0 * 8, &As[0][aRow0][aSeg0 * 8]);
        async_copy_b128(A + (size_t)(mBase + aRow1) * Kd + k0 + aSeg1 * 8, &As[0][aRow1][aSeg1 * 8]);
        async_copy_b128(Bw + (size_t)(k0 + bRow0) * N + nBase + bSeg0 * 8, &Bs[0][bRow0][bSeg0 * 8]);
        async_copy_b128(Bw + (size_t)(k0 + bRow1) * N + nBase + bSeg1 * 8, &Bs[0][bRow1][bSeg1 * 8]);
    }

    for (int kt = 0; kt < nK; ++kt) {
        // tile kt fully in LDS: my async loads done, then block-wide barrier.
        wait_async_all();
        __syncthreads();

        // kick off tile kt+1 into the other buffer (overlaps with WMMA below).
        // Safe: every wave's ds reads of that buffer completed before the
        // barrier above (s_wait_dscnt precedes the wmma consumption).
        if (kt + 1 < nK) {
            const int k1 = (kt + 1) * BK;
            const int nb = buf ^ 1;
            async_copy_b128(A + (size_t)(mBase + aRow0) * Kd + k1 + aSeg0 * 8, &As[nb][aRow0][aSeg0 * 8]);
            async_copy_b128(A + (size_t)(mBase + aRow1) * Kd + k1 + aSeg1 * 8, &As[nb][aRow1][aSeg1 * 8]);
            async_copy_b128(Bw + (size_t)(k1 + bRow0) * N + nBase + bSeg0 * 8, &Bs[nb][bRow0][bSeg0 * 8]);
            async_copy_b128(Bw + (size_t)(k1 + bRow1) * N + nBase + bSeg1 * 8, &Bs[nb][bRow1][bSeg1 * 8]);
        }

        // A fragments: 16x32, lane<16 holds row, K{0..7,16..23}; lane>=16 K{8..15,24..31}
        v16bf af[2];
        #pragma unroll
        for (int m = 0; m < 2; ++m) {
            const int r = waveM * 32 + m * 16 + laneLo;
            Pair p;
            p.lo = *(const U4*)&As[buf][r][laneHi * 8];
            p.hi = *(const U4*)&As[buf][r][16 + laneHi * 8];
            af[m] = __builtin_bit_cast(v16bf, p);
        }
        // B fragments: 32x16, lane<16 holds column, K{0..7,16..23}; lane>=16 K{8..15,24..31}
        v16bf bfr[4];
        #pragma unroll
        for (int n = 0; n < 4; ++n) {
            const int c0 = waveN * 64 + n * 16 + laneLo;
            ushort_t tmp[16];
            #pragma unroll
            for (int s = 0; s < 8; ++s) {
                tmp[s]     = Bs[buf][laneHi * 8 + s][c0];
                tmp[8 + s] = Bs[buf][16 + laneHi * 8 + s][c0];
            }
            bfr[n] = __builtin_bit_cast(v16bf, tmp);
        }
        #pragma unroll
        for (int m = 0; m < 2; ++m)
            #pragma unroll
            for (int n = 0; n < 4; ++n)
                acc[m * 4 + n] = __builtin_amdgcn_wmma_f32_16x16x32_bf16(
                    false, af[m], false, bfr[n], (short)0, acc[m * 4 + n], false, false);

        buf ^= 1;
    }

    // epilogue: C/D layout: VGPR r -> row = r + (lane>=16 ? 8 : 0), col = lane&15
    #pragma unroll
    for (int m = 0; m < 2; ++m) {
        #pragma unroll
        for (int n = 0; n < 4; ++n) {
            v8f c = acc[m * 4 + n];
            const int col = nBase + waveN * 64 + n * 16 + laneLo;
            const float bcol = bias[col];
            #pragma unroll
            for (int r = 0; r < 8; ++r) {
                const int row = mBase + waveM * 32 + m * 16 + r + laneHi * 8;
                const size_t idx = (size_t)row * N + col;
                float val = c[r] + bcol;
                if (mode == 2)       outB[idx] = f2bf(gelu_f(val));
                else if (mode == 1)  outF[idx] = val + resid[idx];
                else                 outF[idx] = val;
            }
        }
    }
}

// ---------------------------------------------------------------------------
// Neighborhood attention: one wave per (b,l,h); lane owns 2 of 64 dims
// ---------------------------------------------------------------------------
__global__ __launch_bounds__(256) void nat_kernel(
    const float* __restrict__ q, const float* __restrict__ k,
    const float* __restrict__ v, const float* __restrict__ rpe,
    ushort_t* __restrict__ att)
{
    const int wid  = blockIdx.x * 8 + (threadIdx.x >> 5);   // 0 .. B*L*H-1
    const int lane = threadIdx.x & 31;
    const int h = wid & 7;
    const int l = (wid >> 3) & 511;
    const int b = wid >> 12;

    int start = l - 3;
    if (start < 0) start = 0;
    if (start > 512 - 7) start = 512 - 7;

    const size_t base = ((size_t)(b * 512 + l) * 512) + h * 64;
    const int d0 = lane * 2;
    const float q0 = q[base + d0], q1 = q[base + d0 + 1];

    float s[7];
    #pragma unroll
    for (int j = 0; j < 7; ++j) {
        const size_t kb = ((size_t)(b * 512 + start + j) * 512) + h * 64 + d0;
        float p = q0 * k[kb] + q1 * k[kb + 1];
        p = wave_sum32(p);                                   // full 64-dim dot
        s[j] = p * 0.125f + rpe[h * 13 + (start + j - l + 6)];
    }
    float mx = s[0];
    #pragma unroll
    for (int j = 1; j < 7; ++j) mx = fmaxf(mx, s[j]);
    float e[7], sum = 0.0f;
    #pragma unroll
    for (int j = 0; j < 7; ++j) { e[j] = __expf(s[j] - mx); sum += e[j]; }
    const float inv = 1.0f / sum;

    float o0 = 0.0f, o1 = 0.0f;
    #pragma unroll
    for (int j = 0; j < 7; ++j) {
        const size_t vb = ((size_t)(b * 512 + start + j) * 512) + h * 64 + d0;
        const float w = e[j] * inv;
        o0 += w * v[vb];
        o1 += w * v[vb + 1];
    }
    unsigned packed = (unsigned)f2bf(o0) | ((unsigned)f2bf(o1) << 16);
    *(unsigned*)&att[base + d0] = packed;
}

// ---------------------------------------------------------------------------
// Host launch
// ---------------------------------------------------------------------------
extern "C" void kernel_launch(void* const* d_in, const int* in_sizes, int n_in,
                              void* d_out, int out_size, void* d_ws, size_t ws_size,
                              hipStream_t stream) {
    const float* x   = (const float*)d_in[0];
    const float* rpe = (const float*)d_in[1];
    const float* g1  = (const float*)d_in[2];
    const float* be1 = (const float*)d_in[3];
    const float* wq  = (const float*)d_in[4];
    const float* bq  = (const float*)d_in[5];
    const float* wk  = (const float*)d_in[6];
    const float* bk  = (const float*)d_in[7];
    const float* wv  = (const float*)d_in[8];
    const float* bv  = (const float*)d_in[9];
    const float* wo  = (const float*)d_in[10];
    const float* bo  = (const float*)d_in[11];
    const float* g2  = (const float*)d_in[12];
    const float* be2 = (const float*)d_in[13];
    const float* w1  = (const float*)d_in[14];
    const float* bm1 = (const float*)d_in[15];
    const float* w2  = (const float*)d_in[16];
    const float* bm2 = (const float*)d_in[17];

    char* ws = (char*)d_ws;
    const size_t MB = 1u << 20;
    ushort_t* wqb = (ushort_t*)(ws + 0);
    ushort_t* wkb = (ushort_t*)(ws + 512 * 1024);
    ushort_t* wvb = (ushort_t*)(ws + 1 * MB);
    ushort_t* wob = (ushort_t*)(ws + 1 * MB + 512 * 1024);
    ushort_t* w1b = (ushort_t*)(ws + 2 * MB);
    ushort_t* w2b = (ushort_t*)(ws + 4 * MB);
    ushort_t* xnb = (ushort_t*)(ws + 6 * MB);    // 16 MB, reused for LN2
    ushort_t* attb = (ushort_t*)(ws + 22 * MB);  // 16 MB
    float*    x2  = (float*)(ws + 38 * MB);      // 32 MB
    float*    qf  = (float*)(ws + 70 * MB);      // 32 MB
    float*    kf  = (float*)(ws + 102 * MB);     // 32 MB
    float*    vf  = (float*)(ws + 134 * MB);     // 32 MB
    ushort_t* h1b = (ushort_t*)(ws + 70 * MB);   // 64 MB, aliases dead q/k

    const int Mrows = 32 * 512;                  // 16384
    float* outp = (float*)d_out;

    // 1) weights -> bf16
    cvt_bf16_kernel<<<1024, 256, 0, stream>>>(wq, wqb, 262144);
    cvt_bf16_kernel<<<1024, 256, 0, stream>>>(wk, wkb, 262144);
    cvt_bf16_kernel<<<1024, 256, 0, stream>>>(wv, wvb, 262144);
    cvt_bf16_kernel<<<1024, 256, 0, stream>>>(wo, wob, 262144);
    cvt_bf16_kernel<<<4096, 256, 0, stream>>>(w1, w1b, 1048576);
    cvt_bf16_kernel<<<4096, 256, 0, stream>>>(w2, w2b, 1048576);

    // 2) LN1
    ln_bf16_kernel<<<Mrows, 256, 0, stream>>>(x, g1, be1, xnb);

    // 3) Q, K, V projections (f32 out for attention)
    gemm_wmma_kernel<<<dim3(128, 4), 256, 0, stream>>>(xnb, wqb, bq, nullptr, qf, nullptr, Mrows, 512, 512, 0);
    gemm_wmma_kernel<<<dim3(128, 4), 256, 0, stream>>>(xnb, wkb, bk, nullptr, kf, nullptr, Mrows, 512, 512, 0);
    gemm_wmma_kernel<<<dim3(128, 4), 256, 0, stream>>>(xnb, wvb, bv, nullptr, vf, nullptr, Mrows, 512, 512, 0);

    // 4) neighborhood attention -> bf16
    nat_kernel<<<16384, 256, 0, stream>>>(qf, kf, vf, rpe, attb);

    // 5) output projection + residual: x2 = x + att @ wo + bo
    gemm_wmma_kernel<<<dim3(128, 4), 256, 0, stream>>>(attb, wob, bo, x, x2, nullptr, Mrows, 512, 512, 1);

    // 6) LN2
    ln_bf16_kernel<<<Mrows, 256, 0, stream>>>(x2, g2, be2, xnb);

    // 7) MLP up + GELU -> bf16
    gemm_wmma_kernel<<<dim3(128, 16), 256, 0, stream>>>(xnb, w1b, bm1, nullptr, nullptr, h1b, Mrows, 2048, 512, 2);

    // 8) MLP down + ORIGINAL-x residual -> out
    gemm_wmma_kernel<<<dim3(128, 4), 256, 0, stream>>>(h1b, w2b, bm2, x, outp, nullptr, Mrows, 512, 2048, 1);

    (void)in_sizes; (void)n_in; (void)out_size; (void)ws_size;
}
